// VectorQuantizer_73177652789912
// MI455X (gfx1250) — compile-verified, matrix operand order checked by reading, not run on hardware
//
#include <hip/hip_runtime.h>
#include <hip/hip_bf16.h>
#include <stdint.h>

// ---------------------------------------------------------------------------
// VQ-VAE vector quantizer for MI455X (gfx1250).
//   z   : [16,256,32,32] f32      emb : [8192,256] f32
//   out : z_q (NCHW, 4194304 f32) ++ idx (16384, stored as f32) ++ loss (1 f32)
//
// Distance GEMM done with V_WMMA_F32_16X16X32_BF16 using a bf16 hi/lo split of
// both operands (3 WMMAs per K-chunk => ~fp32-accurate dot products).
// Argmin is fused into registers; codebook tiles are double-buffered in LDS
// via async global->LDS copies (global_load_async_to_lds_b128).
// ---------------------------------------------------------------------------

typedef __attribute__((ext_vector_type(16))) __bf16          v16bf;
typedef __attribute__((ext_vector_type(16))) unsigned short  v16u;
typedef __attribute__((ext_vector_type(8)))  float           v8f;
typedef __attribute__((ext_vector_type(4)))  int             v4i;

#define N_ROWS   16384   // 16*32*32
#define N_CODES  8192
#define CDIM     256
#define SPLITS   4       // code-range splits for K1 parallelism
#define TILESPS  128     // (8192/4)/16 code tiles per split

#if defined(__AMDGCN__) && __has_builtin(__builtin_amdgcn_global_load_async_to_lds_b128)
#define USE_ASYNC_LDS 1
// exact parameter types per hipcc diagnostic: (v4i __device__*, v4i __shared__*, imm, imm)
typedef __attribute__((address_space(1))) v4i* gptr_v4i;
typedef __attribute__((address_space(3))) v4i* lptr_v4i;
#else
#define USE_ASYNC_LDS 0
#endif

__device__ __forceinline__ void bf16_split(float x, unsigned short& h, unsigned short& l) {
    unsigned u = __float_as_uint(x);
    h = (unsigned short)(u >> 16);                       // truncate to bf16
    float rem = x - __uint_as_float((unsigned)h << 16);  // exact residual
    l = (unsigned short)(__float_as_uint(rem) >> 16);
}

__device__ __forceinline__ float blockReduceSum(float v) {
    __shared__ float smem[8];
    int lane = threadIdx.x & 31, w = threadIdx.x >> 5;
#pragma unroll
    for (int m = 16; m; m >>= 1) v += __shfl_xor(v, m, 32);
    if (lane == 0) smem[w] = v;
    __syncthreads();
    if (w == 0) {
        v = (lane < (int)(blockDim.x >> 5)) ? smem[lane] : 0.0f;
#pragma unroll
        for (int m = 4; m; m >>= 1) v += __shfl_xor(v, m, 32);
    }
    return v;  // valid in thread 0
}

// ---- K0: emb -> bf16 hi/lo split + ||e||^2 --------------------------------
__global__ __launch_bounds__(256)
void vq_prep_emb_kernel(const float* __restrict__ emb,
                        unsigned short* __restrict__ ehi,
                        unsigned short* __restrict__ elo,
                        float* __restrict__ enorm) {
    int k = blockIdx.x;     // code row
    int c = threadIdx.x;    // channel
    float x = emb[(size_t)k * CDIM + c];
    unsigned short h, l;
    bf16_split(x, h, l);
    ehi[(size_t)k * CDIM + c] = h;
    elo[(size_t)k * CDIM + c] = l;
    float s = blockReduceSum(x * x);
    if (threadIdx.x == 0) enorm[k] = s;
}

// ---- K1: fused distance GEMM + argmin -------------------------------------
// One 16-code tile in LDS = [2 parts][16 codes][256 k] bf16 = 16KB, x2 buffers.
__device__ __forceinline__ void issue_tile_copy(const unsigned short* __restrict__ ehi,
                                                const unsigned short* __restrict__ elo,
                                                unsigned short* sbuf,   // ebuf + buf*8192
                                                int codebase, int tid) {
    // 1024 chunks of 16B (both parts); 256 threads x 4 chunks
#pragma unroll
    for (int i = 0; i < 4; ++i) {
        int chunk = tid + (i << 8);
        int part  = chunk >> 9;
        int co    = (chunk & 511) << 3;   // ushort offset within part
        const unsigned short* src = (part ? elo : ehi) + (size_t)codebase * CDIM + co;
        unsigned short*       dst = sbuf + part * 4096 + co;
#if USE_ASYNC_LDS
        __builtin_amdgcn_global_load_async_to_lds_b128((gptr_v4i)src, (lptr_v4i)dst, 0, 0);
#else
        *(uint4*)dst = *(const uint4*)src;
#endif
    }
}

__device__ __forceinline__ void wait_async_copies() {
#if USE_ASYNC_LDS
#if __has_builtin(__builtin_amdgcn_s_wait_asynccnt)
    __builtin_amdgcn_s_wait_asynccnt(0);
#else
    asm volatile("s_wait_asynccnt 0" ::: "memory");
#endif
#endif
}

__global__ __launch_bounds__(256)
void vq_dist_argmin_kernel(const float* __restrict__ z,
                           const unsigned short* __restrict__ ehi,
                           const unsigned short* __restrict__ elo,
                           const float* __restrict__ enorm,
                           float* __restrict__ cand_val,
                           int*   __restrict__ cand_idx) {
    __shared__ __align__(32) unsigned short ebuf[2 * 2 * 16 * CDIM];  // 32 KB

    const int tid  = threadIdx.x;
    const int lane = tid & 31;
    const int wave = tid >> 5;
    const int rowTile = blockIdx.x & 127;   // 128 row tiles of 128 rows
    const int split   = blockIdx.x >> 7;    // 0..3 code ranges of 2048
    const int rowbase = rowTile * 128 + wave * 16;
    const int hv  = lane >> 4;              // half-wave (WMMA K split)
    const int l15 = lane & 15;

    // ---- A operand: this wave's 16 z-rows, split to bf16 hi/lo in VGPRs ----
    // z_flat[r][k] = z[b][k][hw] with b=r>>10, hw=r&1023 (stride 1024 in k).
    // A k order used here (contiguous, B matches): lanes0-15 row=l15 K 0..15, lanes16-31 row=l15 K 16..31.
    const int r    = rowbase + l15;
    const int bimg = r >> 10;
    const int hw   = r & 1023;
    const float* zrow = z + (size_t)bimg * (CDIM * 1024) + hw;
    const int kofs = hv << 4;

    v16u Ahi[8], Alo[8];
#pragma unroll
    for (int c = 0; c < 8; ++c) {
#pragma unroll
        for (int j = 0; j < 16; ++j) {
            float x = zrow[(size_t)(c * 32 + kofs + j) << 10];
            unsigned short h, l;
            bf16_split(x, h, l);
            Ahi[c][j] = h;
            Alo[c][j] = l;
        }
    }

    float bestV[8];
    int   bestI[8];
#pragma unroll
    for (int v = 0; v < 8; ++v) { bestV[v] = 3.4e38f; bestI[v] = 0; }

    const int code0 = split * (N_CODES / SPLITS);

    issue_tile_copy(ehi, elo, ebuf, code0, tid);
    wait_async_copies();
    __syncthreads();

    int cur = 0;
    for (int t = 0; t < TILESPS; ++t) {
        if (t + 1 < TILESPS)
            issue_tile_copy(ehi, elo, ebuf + (cur ^ 1) * 8192, code0 + (t + 1) * 16, tid);

        // B operand: code=l15's row of k values; half-wave selects K 0..15/16..31.
        const unsigned short* bb = ebuf + cur * 8192 + l15 * CDIM + (hv << 4);

        v8f acc0 = {};
        v8f acc1 = {};
#pragma unroll
        for (int c = 0; c < 8; ++c) {
            v16bf bh = __builtin_bit_cast(v16bf, *(const v16u*)(bb + c * 32));
            v16bf bl = __builtin_bit_cast(v16bf, *(const v16u*)(bb + 4096 + c * 32));
            v16bf ah = __builtin_bit_cast(v16bf, Ahi[c]);
            v16bf al = __builtin_bit_cast(v16bf, Alo[c]);
            acc0 = __builtin_amdgcn_wmma_f32_16x16x32_bf16(false, ah, false, bh, (short)0, acc0, false, false);
            acc1 = __builtin_amdgcn_wmma_f32_16x16x32_bf16(false, ah, false, bl, (short)0, acc1, false, false);
            acc1 = __builtin_amdgcn_wmma_f32_16x16x32_bf16(false, al, false, bh, (short)0, acc1, false, false);
        }

        // score = ||e||^2 - 2*z.e   (||z||^2 is row-constant, irrelevant to argmin)
        const int   col = code0 + t * 16 + l15;
        const float en  = enorm[col];
#pragma unroll
        for (int v = 0; v < 8; ++v) {
            float s = en - 2.0f * (acc0[v] + acc1[v]);
            if (s < bestV[v]) { bestV[v] = s; bestI[v] = col; }  // strict < keeps first idx
        }

        wait_async_copies();
        __syncthreads();
        cur ^= 1;
    }

    // C/D layout: lane group {0-15|16-31} = col, VGPR v = row v (+8 for hi half).
    // Reduce the 16 candidate columns per row across each half-wave.
#pragma unroll
    for (int v = 0; v < 8; ++v) {
        float bv = bestV[v];
        int   bi = bestI[v];
#pragma unroll
        for (int m = 8; m; m >>= 1) {
            float ov = __shfl_xor(bv, m, 16);
            int   oi = __shfl_xor(bi, m, 16);
            if (ov < bv || (ov == bv && oi < bi)) { bv = ov; bi = oi; }
        }
        if (l15 == 0) {
            int row = rowbase + v + (hv << 3);
            cand_val[split * N_ROWS + row] = bv;
            cand_idx[split * N_ROWS + row] = bi;
        }
    }
}

// ---- K2: merge split candidates -------------------------------------------
__global__ __launch_bounds__(256)
void vq_combine_kernel(const float* __restrict__ cand_val,
                       const int*   __restrict__ cand_idx,
                       int*   __restrict__ idx_int,
                       float* __restrict__ idx_out) {
    int n = blockIdx.x * blockDim.x + threadIdx.x;
    float bv = cand_val[n];
    int   bi = cand_idx[n];
#pragma unroll
    for (int s = 1; s < SPLITS; ++s) {
        float v = cand_val[s * N_ROWS + n];
        int   i = cand_idx[s * N_ROWS + n];
        if (v < bv) { bv = v; bi = i; }   // splits ascend in code idx: ties keep first
    }
    idx_int[n] = bi;
    idx_out[n] = (float)bi;
}

// ---- K3: gather z_q, write NCHW output, per-row loss partial ---------------
__global__ __launch_bounds__(256)
void vq_gather_out_kernel(const float* __restrict__ z,
                          const float* __restrict__ emb,
                          const int*   __restrict__ idx,
                          float* __restrict__ out,
                          float* __restrict__ partial) {
    int n = blockIdx.x;     // flattened (b,h,w)
    int c = threadIdx.x;    // channel
    int id = idx[n];
    int bimg = n >> 10, hw = n & 1023;
    size_t zo = ((size_t)bimg * CDIM + c) * 1024 + hw;   // NCHW offset
    float ev = emb[(size_t)id * CDIM + c];
    float zv = z[zo];
    out[zo] = ev;  // straight-through value == z_q numerically
    float d = ev - zv;
    float s = blockReduceSum(d * d);
    if (c == 0) partial[n] = s;
}

// ---- K4: loss = (1 + beta) * mean((z_q - z)^2) -----------------------------
__global__ __launch_bounds__(256)
void vq_loss_kernel(const float* __restrict__ partial, float* __restrict__ loss_out) {
    float s = 0.0f;
    for (int i = threadIdx.x; i < N_ROWS; i += 256) s += partial[i];
    s = blockReduceSum(s);
    if (threadIdx.x == 0)
        loss_out[0] = 1.25f * s / ((float)N_ROWS * (float)CDIM);
}

// ---------------------------------------------------------------------------
extern "C" void kernel_launch(void* const* d_in, const int* in_sizes, int n_in,
                              void* d_out, int out_size, void* d_ws, size_t ws_size,
                              hipStream_t stream) {
    const float* z   = (const float*)d_in[0];   // [16,256,32,32]
    const float* emb = (const float*)d_in[1];   // [8192,256]

    char* ws = (char*)d_ws;
    unsigned short* ehi      = (unsigned short*)(ws + 0);          // 4 MB
    unsigned short* elo      = (unsigned short*)(ws + 4194304);    // 4 MB
    float*          enorm    = (float*)(ws + 8388608);             // 32 KB
    float*          cand_val = (float*)(ws + 8421376);             // 256 KB
    int*            cand_idx = (int*)  (ws + 8683520);             // 256 KB
    int*            idx_int  = (int*)  (ws + 8945664);             // 64 KB
    float*          partial  = (float*)(ws + 9011200);             // 64 KB

    float* out_zq   = (float*)d_out;            // 4194304
    float* out_idx  = out_zq + 4194304;         // 16384 (idx as float)
    float* out_loss = out_idx + N_ROWS;         // 1

    vq_prep_emb_kernel  <<<N_CODES, 256, 0, stream>>>(emb, ehi, elo, enorm);
    vq_dist_argmin_kernel<<<128 * SPLITS, 256, 0, stream>>>(z, ehi, elo, enorm,
                                                            cand_val, cand_idx);
    vq_combine_kernel   <<<N_ROWS / 256, 256, 0, stream>>>(cand_val, cand_idx,
                                                           idx_int, out_idx);
    vq_gather_out_kernel<<<N_ROWS, 256, 0, stream>>>(z, emb, idx_int, out_zq, partial);
    vq_loss_kernel      <<<1, 256, 0, stream>>>(partial, out_loss);
}
